// LinearPredictor_16578573762946
// MI455X (gfx1250) — matrix operand order repaired and checked by reading, not compiled, not run on hardware
//
#include <hip/hip_runtime.h>
#include <stddef.h>
#include <stdint.h>

// ---------------- model dims ----------------
#define NPED 2048
#define TT   8
#define EE   64
#define HID  1024
#define KS   20          // num_samples
#define PRED 12
#define PROJ_N (HID*KS)  // 20480
#define PROJ_K (EE*TT)   // 512
#define DD   256
#define M2   (NPED*KS)   // 40960 rows for dr GEMM

typedef unsigned short ushort_t;
typedef unsigned int   uint_t;

typedef __attribute__((ext_vector_type(16))) __bf16 v16bf;
typedef __attribute__((ext_vector_type(8)))  float  v8f;
typedef __attribute__((ext_vector_type(4)))  unsigned int u32x4;
typedef __attribute__((ext_vector_type(4)))  int  i32x4;
typedef __attribute__((ext_vector_type(8)))  int  i32x8;

union FragBF { uint_t u[8]; v16bf v; };

// fp32 -> bf16 round-to-nearest-even
__device__ __forceinline__ ushort_t f2bf(float f) {
  uint_t u = __float_as_uint(f);
  u += 0x7FFFu + ((u >> 16) & 1u);
  return (ushort_t)(u >> 16);
}
__device__ __forceinline__ float bf2f(ushort_t h) {
  return __uint_as_float(((uint_t)h) << 16);
}
__device__ __forceinline__ float wave_sum(float v) {
#pragma unroll
  for (int o = 16; o > 0; o >>= 1) v += __shfl_xor(v, o, 32);
  return v;
}
__device__ __forceinline__ float sigm(float x) { return 1.f / (1.f + __expf(-x)); }

// ---------------------------------------------------------------------------
// Tensor Data Mover: 2-D tile (4-byte elements) global -> LDS.
// D# group0: [1:0]=count=1, [63:32]=lds_addr, [120:64]=global_addr, [127:126]=type=2
// D# group1: [17:16]=data_size(2=4B), dim0/dim1, tile0/tile1, dim0_stride
// ---------------------------------------------------------------------------
__device__ __forceinline__ void tdm_load_2d(const void* gsrc, unsigned lds_addr,
                                            unsigned tile_d0, unsigned tile_d1,
                                            unsigned tensor_d0, unsigned tensor_d1,
                                            unsigned long long stride_d0) {
  unsigned long long ga = (unsigned long long)(uintptr_t)gsrc;
  u32x4 g0 = { 1u,
               lds_addr,
               (unsigned)(ga & 0xFFFFFFFFull),
               (unsigned)((ga >> 32) & 0x01FFFFFFull) | (2u << 30) };
  i32x8 g1 = { (int)(2u << 16),                                        // data_size=4B
               (int)((tensor_d0 & 0xFFFFu) << 16),                     // dim0 lo16
               (int)(((tensor_d0 >> 16) & 0xFFFFu) | ((tensor_d1 & 0xFFFFu) << 16)),
               (int)(((tensor_d1 >> 16) & 0xFFFFu) | ((tile_d0 & 0xFFFFu) << 16)),
               (int)(tile_d1 & 0xFFFFu),                               // tile_dim1, tile_dim2=0
               (int)(unsigned)(stride_d0 & 0xFFFFFFFFull),             // dim0 stride lo32
               (int)(unsigned)((stride_d0 >> 32) & 0xFFFFull),         // dim0 stride hi16
               0 };
  i32x4 z4 = { 0, 0, 0, 0 };
#if __clang_major__ >= 23
  i32x8 z8 = { 0, 0, 0, 0, 0, 0, 0, 0 };
  __builtin_amdgcn_tensor_load_to_lds(g0, g1, z4, z4, z8, 0);
#else
  __builtin_amdgcn_tensor_load_to_lds(g0, g1, z4, z4, 0);
#endif
}
__device__ __forceinline__ unsigned lds_off(const void* p) {
  return (unsigned)(uintptr_t)p;       // generic LDS pointer: low 32 bits = LDS byte addr
}

// ---------------------------------------------------------------------------
// Pack f32 weight [K,N] into WMMA-fragment-contiguous bf16 tiles:
// BT[tk][tn][lane][v] (u32 pair words), lane = g*16+nB, pair = v + g*8.
// A lane's 8 B-fragment words are 32 contiguous bytes -> 2x b128 per k-step.
// ---------------------------------------------------------------------------
__global__ void pack_w_tiles(const float* __restrict__ W, uint_t* __restrict__ out,
                             int K, int N) {
  int idx = blockIdx.x * blockDim.x + threadIdx.x;
  int tot = (K >> 1) * N;
  if (idx >= tot) return;
  int v    = idx & 7;
  int lane = (idx >> 3) & 31;
  int rest = idx >> 8;
  int ntiles = N >> 4;
  int tn = rest % ntiles, tk = rest / ntiles;
  int k2 = tk * 16 + (v + ((lane >> 4) << 3));
  int n  = tn * 16 + (lane & 15);
  ushort_t lo = f2bf(W[(2 * k2) * N + n]);
  ushort_t hi = f2bf(W[(2 * k2 + 1) * N + n]);
  out[idx] = (uint_t)lo | ((uint_t)hi << 16);
}

// ---------------------------------------------------------------------------
// One wave32 per pedestrian: embedding+PE, 3 post-norm encoder layers, GRU.
// Emits A matrix [NPED, 512] row-major bf16 (== packed WMMA A pair words).
// ---------------------------------------------------------------------------
__global__ __launch_bounds__(32) void encode_gru(
    const float* __restrict__ x,
    const float* __restrict__ emb_w, const float* __restrict__ emb_b,
    const float* __restrict__ qw, const float* __restrict__ qb,
    const float* __restrict__ kw, const float* __restrict__ kb,
    const float* __restrict__ vw, const float* __restrict__ vb,
    const float* __restrict__ ow, const float* __restrict__ ob,
    const float* __restrict__ f1w, const float* __restrict__ f1b,
    const float* __restrict__ f2w, const float* __restrict__ f2b,
    const float* __restrict__ ln1g, const float* __restrict__ ln1b,
    const float* __restrict__ ln2g, const float* __restrict__ ln2b,
    const float* __restrict__ gih, const float* __restrict__ ghh,
    const float* __restrict__ gbih, const float* __restrict__ gbhh,
    ushort_t* __restrict__ Aout)
{
  __shared__ float sh_h[TT][EE];
  __shared__ float sh_q[TT][EE];
  __shared__ float sh_k[TT][EE];
  __shared__ float sh_v[TT][EE];
  __shared__ float sh_o[TT][EE];
  __shared__ float sh_xg[TT][3 * EE];
  __shared__ float sh_hs[EE];

  const int n = blockIdx.x;
  const int t = threadIdx.x;

  for (int s = 0; s < TT; ++s) {
    float x0 = x[(n * TT + s) * 2 + 0];
    float x1 = x[(n * TT + s) * 2 + 1];
#pragma unroll
    for (int rep = 0; rep < 2; ++rep) {
      int c = t + rep * 32;
      float e = x0 * emb_w[c] + x1 * emb_w[EE + c] + emb_b[c];
      e = e > 0.f ? e : 0.f;
      int i2 = (c >> 1) * 2;
      float div = __expf((float)i2 * (-9.210340372f / (float)EE));
      float ang = (float)s * div;
      e += (c & 1) ? __cosf(ang) : __sinf(ang);
      sh_h[s][c] = e;
    }
  }
  __syncthreads();

  for (int L = 0; L < 3; ++L) {
    const float *Wq = qw + L * EE * EE, *Bq = qb + L * EE;
    const float *Wk = kw + L * EE * EE, *Bk = kb + L * EE;
    const float *Wv = vw + L * EE * EE, *Bv = vb + L * EE;
    const float *Wo = ow + L * EE * EE, *Bo = ob + L * EE;
    const float *W1 = f1w + L * EE * EE, *B1 = f1b + L * EE;
    const float *W2 = f2w + L * EE * EE, *B2 = f2b + L * EE;
    const float *G1 = ln1g + L * EE, *Be1 = ln1b + L * EE;
    const float *G2 = ln2g + L * EE, *Be2 = ln2b + L * EE;

    for (int s = 0; s < TT; ++s) {
#pragma unroll
      for (int rep = 0; rep < 2; ++rep) {
        int c = t + rep * 32;
        float aq = Bq[c], ak = Bk[c], av = Bv[c];
        for (int k = 0; k < EE; ++k) {
          float hv = sh_h[s][k];
          aq += hv * Wq[k * EE + c];
          ak += hv * Wk[k * EE + c];
          av += hv * Wv[k * EE + c];
        }
        sh_q[s][c] = aq; sh_k[s][c] = ak; sh_v[s][c] = av;
      }
    }
    __syncthreads();

#pragma unroll
    for (int rep = 0; rep < 2; ++rep) {
      int id = t + rep * 32;
      int hd = id >> 3, s = id & 7;
      float lg[TT], mx = -1e30f;
#pragma unroll
      for (int s2 = 0; s2 < TT; ++s2) {
        float d = 0.f;
#pragma unroll
        for (int dd = 0; dd < 8; ++dd)
          d += sh_q[s][hd * 8 + dd] * sh_k[s2][hd * 8 + dd];
        d *= 0.3535533906f;
        lg[s2] = d; mx = d > mx ? d : mx;
      }
      float den = 0.f;
#pragma unroll
      for (int s2 = 0; s2 < TT; ++s2) { lg[s2] = __expf(lg[s2] - mx); den += lg[s2]; }
      float inv = 1.f / den;
#pragma unroll
      for (int dd = 0; dd < 8; ++dd) {
        float o = 0.f;
#pragma unroll
        for (int s2 = 0; s2 < TT; ++s2) o += lg[s2] * sh_v[s2][hd * 8 + dd];
        sh_o[s][hd * 8 + dd] = o * inv;
      }
    }
    __syncthreads();

    for (int s = 0; s < TT; ++s) {
#pragma unroll
      for (int rep = 0; rep < 2; ++rep) {
        int c = t + rep * 32;
        float a = Bo[c];
        for (int k = 0; k < EE; ++k) a += sh_o[s][k] * Wo[k * EE + c];
        sh_q[s][c] = sh_h[s][c] + a;
      }
    }
    __syncthreads();
    for (int s = 0; s < TT; ++s) {
      float a0 = sh_q[s][t], a1 = sh_q[s][t + 32];
      float m = wave_sum(a0 + a1) * (1.f / 64.f);
      float d0 = a0 - m, d1 = a1 - m;
      float var = wave_sum(d0 * d0 + d1 * d1) * (1.f / 64.f);
      float rs = rsqrtf(var + 1e-5f);
      sh_h[s][t] = d0 * rs * G1[t] + Be1[t];
      sh_h[s][t + 32] = d1 * rs * G1[t + 32] + Be1[t + 32];
    }
    __syncthreads();

    for (int s = 0; s < TT; ++s) {
#pragma unroll
      for (int rep = 0; rep < 2; ++rep) {
        int c = t + rep * 32;
        float a = B1[c];
        for (int k = 0; k < EE; ++k) a += sh_h[s][k] * W1[k * EE + c];
        sh_q[s][c] = a > 0.f ? a : 0.f;
      }
    }
    __syncthreads();
    for (int s = 0; s < TT; ++s) {
#pragma unroll
      for (int rep = 0; rep < 2; ++rep) {
        int c = t + rep * 32;
        float a = B2[c];
        for (int k = 0; k < EE; ++k) a += sh_q[s][k] * W2[k * EE + c];
        sh_k[s][c] = sh_h[s][c] + a;
      }
    }
    __syncthreads();
    for (int s = 0; s < TT; ++s) {
      float a0 = sh_k[s][t], a1 = sh_k[s][t + 32];
      float m = wave_sum(a0 + a1) * (1.f / 64.f);
      float d0 = a0 - m, d1 = a1 - m;
      float var = wave_sum(d0 * d0 + d1 * d1) * (1.f / 64.f);
      float rs = rsqrtf(var + 1e-5f);
      sh_h[s][t] = d0 * rs * G2[t] + Be2[t];
      sh_h[s][t + 32] = d1 * rs * G2[t + 32] + Be2[t + 32];
    }
    __syncthreads();
  }

  for (int s = 0; s < TT; ++s) {
#pragma unroll
    for (int j = 0; j < 6; ++j) {
      int c3 = t + j * 32;
      float a = gbih[c3];
      for (int k = 0; k < EE; ++k) a += sh_h[s][k] * gih[k * 192 + c3];
      sh_xg[s][c3] = a;
    }
  }
  sh_hs[t] = 0.f; sh_hs[t + 32] = 0.f;
  __syncthreads();

  for (int s = 0; s < TT; ++s) {
    float hg[6];
#pragma unroll
    for (int j = 0; j < 6; ++j) {
      int c3 = t + j * 32;
      float a = gbhh[c3];
      for (int k = 0; k < EE; ++k) a += sh_hs[k] * ghh[k * 192 + c3];
      hg[j] = a;
    }
    __syncthreads();
#pragma unroll
    for (int rep = 0; rep < 2; ++rep) {
      int c = t + rep * 32;
      float r = sigm(sh_xg[s][c] + hg[0 + rep]);
      float z = sigm(sh_xg[s][64 + c] + hg[2 + rep]);
      float nn = tanhf(sh_xg[s][128 + c] + r * hg[4 + rep]);
      float hnew = (1.f - z) * nn + z * sh_hs[c];
      sh_hs[c] = hnew;
      Aout[(size_t)n * PROJ_K + s * EE + c] = f2bf(hnew);
    }
    __syncthreads();
  }
}

// ---------------------------------------------------------------------------
// bf16 WMMA GEMM, fp32 accumulate, TDM double-buffered staging.
//   A: row-major bf16 pair words [M, Kt/2]   (2-D strided TDM tile, 16x16 u32)
//   B: fragment-contiguous tiles BT[tk][tn][lane][v]  (1-D TDM chunk, 16 KB)
// Block = 256 threads = 8 waves; block tile = 16 M x 256 N; each wave 16x32.
// Wave 0 drives the Tensor Data Mover; s_wait_tensorcnt(2) keeps one tile
// in flight behind the tile being computed (in-order TDM completion).
// EPI 0: raw f32 out (LayerNorm follows).  EPI 1: bias+PReLU -> bf16.
// ---------------------------------------------------------------------------
template <int EPI, int Nt, int Kt>
__global__ __launch_bounds__(256) void gemm_wmma(
    const uint_t* __restrict__ A32, const uint_t* __restrict__ BT,
    const float* __restrict__ bias,
    float* __restrict__ Cf, ushort_t* __restrict__ Cbf,
    const float* __restrict__ prelu_a)
{
  constexpr int K2 = Kt / 2;
  constexpr int nT = K2 / 16;          // k-tiles of 32 K
  constexpr int NSUB = 16;             // 16 n-subtiles = 256 N per block
  __shared__ uint_t lA[2][16 * 16];
  __shared__ uint_t lB[2][NSUB * 256];

  const int lane = threadIdx.x & 31;
  const int wave = threadIdx.x >> 5;
  const int m0 = blockIdx.y * 16;
  const int n0blk = blockIdx.x * 256;
  const int g = lane >> 4, nB = lane & 15, mA = lane & 15;

  auto issue = [&](int i) {
    int buf = i & 1;
    // A tile: 16 rows x 16 pair-words, row stride K2
    tdm_load_2d(A32 + (size_t)m0 * K2 + i * 16, lds_off(&lA[buf][0]),
                16, 16, 16, 16, (unsigned long long)K2);
    // B chunk: contiguous NSUB*256 u32
    const uint_t* bsrc = BT + ((size_t)i * (Nt / 16) + (n0blk / 16)) * 256;
    tdm_load_2d(bsrc, lds_off(&lB[buf][0]),
                NSUB * 256, 1, NSUB * 256, 1, (unsigned long long)(NSUB * 256));
  };

  if (wave == 0) { issue(0); if (nT > 1) issue(1); }

  v8f c0 = {}, c1 = {};
  for (int i = 0; i < nT; ++i) {
    if (wave == 0) {
      if (i + 1 < nT) __builtin_amdgcn_s_wait_tensorcnt(2);  // tile i resident
      else            __builtin_amdgcn_s_wait_tensorcnt(0);
    }
    __syncthreads();
    const uint_t* A_ = &lA[i & 1][0];
    const uint_t* B_ = &lB[i & 1][0];
    FragBF a, b0, b1;
#pragma unroll
    for (int v = 0; v < 8; ++v) {
      int pa = v + (g << 3);                                  // contiguous A pair map (matches B pack)
      a.u[v]  = A_[mA * 16 + pa];
      b0.u[v] = B_[(wave * 2 + 0) * 256 + lane * 8 + v];      // contiguous 32B
      b1.u[v] = B_[(wave * 2 + 1) * 256 + lane * 8 + v];
    }
    c0 = __builtin_amdgcn_wmma_f32_16x16x32_bf16(false, a.v, false, b0.v,
                                                 (short)0, c0, false, false);
    c1 = __builtin_amdgcn_wmma_f32_16x16x32_bf16(false, a.v, false, b1.v,
                                                 (short)0, c1, false, false);
    __syncthreads();
    if (wave == 0 && i + 2 < nT) issue(i + 2);                // buf freed by barrier
  }

  const float pa_ = (EPI == 1) ? prelu_a[0] : 0.f;
#pragma unroll
  for (int j = 0; j < 2; ++j) {
    const v8f& c = j ? c1 : c0;
    int n = n0blk + (wave * 2 + j) * 16 + nB;
    float bn = bias[n];
#pragma unroll
    for (int v = 0; v < 8; ++v) {
      int m = m0 + v + (g << 3);                              // ISA f32 C/D map
      float val = c[v] + bn;
      if (EPI == 0) {
        Cf[(size_t)m * Nt + n] = val;
      } else {
        float y = val >= 0.f ? val : pa_ * val;
        Cbf[(size_t)m * Nt + n] = f2bf(y);
      }
    }
  }
}

// ---------------------------------------------------------------------------
__global__ __launch_bounds__(256) void ln_relu_rows(
    const float* __restrict__ Z, const float* __restrict__ g,
    const float* __restrict__ b, ushort_t* __restrict__ Y, int Ncols)
{
  const int row = blockIdx.x;
  const float* z = Z + (size_t)row * Ncols;
  __shared__ float red[2][8];
  float s = 0.f, s2 = 0.f;
  for (int cidx = threadIdx.x; cidx < Ncols; cidx += 256) {
    float v = z[cidx]; s += v; s2 += v * v;
  }
  s = wave_sum(s); s2 = wave_sum(s2);
  if ((threadIdx.x & 31) == 0) { red[0][threadIdx.x >> 5] = s; red[1][threadIdx.x >> 5] = s2; }
  __syncthreads();
  float ts = 0.f, ts2 = 0.f;
#pragma unroll
  for (int w = 0; w < 8; ++w) { ts += red[0][w]; ts2 += red[1][w]; }
  float mean = ts / (float)Ncols;
  float var = ts2 / (float)Ncols - mean * mean;
  float rstd = rsqrtf(var + 1e-5f);
  ushort_t* y = Y + (size_t)row * Ncols;
  for (int cidx = threadIdx.x; cidx < Ncols; cidx += 256) {
    float v = (z[cidx] - mean) * rstd * g[cidx] + b[cidx];
    y[cidx] = f2bf(v > 0.f ? v : 0.f);
  }
}

// ---------------------------------------------------------------------------
__global__ void out_head(const ushort_t* __restrict__ lat2,
                         const float* __restrict__ Wo, const float* __restrict__ bo,
                         float* __restrict__ out)
{
  int idx = blockIdx.x * blockDim.x + threadIdx.x;
  const int TOTN = M2 * 24;
  if (idx >= TOTN) return;
  int row = idx / 24, col = idx - (idx / 24) * 24;
  const ushort_t* a = lat2 + (size_t)row * DD;
  float acc = bo[col];
  for (int d = 0; d < DD; ++d) acc += bf2f(a[d]) * Wo[d * 24 + col];
  int n = row / KS, kk = row - n * KS;
  int pr = col >> 1, o = col & 1;
  out[(((size_t)kk * NPED + n) * PRED + pr) * 2 + o] = acc;
}

// ---------------------------------------------------------------------------
extern "C" void kernel_launch(void* const* d_in, const int* in_sizes, int n_in,
                              void* d_out, int out_size, void* d_ws, size_t ws_size,
                              hipStream_t stream) {
  (void)in_sizes; (void)n_in; (void)out_size; (void)ws_size;
  const float* x      = (const float*)d_in[0];
  const float* emb_w  = (const float*)d_in[4];
  const float* emb_b  = (const float*)d_in[5];
  const float* eqw = (const float*)d_in[24]; const float* eqb = (const float*)d_in[25];
  const float* ekw = (const float*)d_in[26]; const float* ekb = (const float*)d_in[27];
  const float* evw = (const float*)d_in[28]; const float* evb = (const float*)d_in[29];
  const float* eow = (const float*)d_in[30]; const float* eob = (const float*)d_in[31];
  const float* e1w = (const float*)d_in[32]; const float* e1b = (const float*)d_in[33];
  const float* e2w = (const float*)d_in[34]; const float* e2b = (const float*)d_in[35];
  const float* l1g = (const float*)d_in[36]; const float* l1b = (const float*)d_in[37];
  const float* l2g = (const float*)d_in[38]; const float* l2b = (const float*)d_in[39];
  const float* gih  = (const float*)d_in[40]; const float* ghh  = (const float*)d_in[41];
  const float* gbih = (const float*)d_in[42]; const float* gbhh = (const float*)d_in[43];
  const float* proj_w = (const float*)d_in[44]; const float* proj_b = (const float*)d_in[45];
  const float* proj_g = (const float*)d_in[46]; const float* proj_be = (const float*)d_in[47];
  const float* dr_w = (const float*)d_in[48]; const float* dr_b = (const float*)d_in[49];
  const float* prelu_a = (const float*)d_in[50];
  const float* ow = (const float*)d_in[77]; const float* obb = (const float*)d_in[78];

  uintptr_t base = (uintptr_t)d_ws;
  size_t off = 0;
  auto carve = [&](size_t bytes) -> void* {
    void* r = (void*)(base + off);
    off = (off + bytes + 255) & ~(size_t)255;
    return r;
  };
  ushort_t* A1  = (ushort_t*)carve((size_t)NPED * PROJ_K * 2);
  uint_t*   Wp  = (uint_t*)  carve((size_t)(PROJ_K / 2) * PROJ_N * 4);
  uint_t*   Wd  = (uint_t*)  carve((size_t)(HID / 2) * DD * 4);
  float*    Z1  = (float*)   carve((size_t)NPED * PROJ_N * 4);
  ushort_t* LAT = (ushort_t*)carve((size_t)NPED * PROJ_N * 2);
  ushort_t* LT2 = (ushort_t*)carve((size_t)M2 * DD * 2);

  {
    int tot = (PROJ_K / 2) * PROJ_N;
    pack_w_tiles<<<(tot + 255) / 256, 256, 0, stream>>>(proj_w, Wp, PROJ_K, PROJ_N);
    tot = (HID / 2) * DD;
    pack_w_tiles<<<(tot + 255) / 256, 256, 0, stream>>>(dr_w, Wd, HID, DD);
  }
  encode_gru<<<NPED, 32, 0, stream>>>(x, emb_w, emb_b,
      eqw, eqb, ekw, ekb, evw, evb, eow, eob, e1w, e1b, e2w, e2b,
      l1g, l1b, l2g, l2b, gih, ghh, gbih, gbhh, A1);
  gemm_wmma<0, PROJ_N, PROJ_K><<<dim3(PROJ_N / 256, NPED / 16), 256, 0, stream>>>(
      (const uint_t*)A1, Wp, proj_b, Z1, nullptr, nullptr);
  ln_relu_rows<<<NPED, 256, 0, stream>>>(Z1, proj_g, proj_be, LAT, PROJ_N);
  gemm_wmma<1, DD, HID><<<dim3(DD / 256, M2 / 16), 256, 0, stream>>>(
      (const uint_t*)LAT, Wd, dr_b, nullptr, LT2, prelu_a);
  {
    int tot = M2 * 24;
    out_head<<<(tot + 255) / 256, 256, 0, stream>>>(LT2, ow, obb, (float*)d_out);
  }
}
